// HeatDGM_49976239456922
// MI455X (gfx1250) — matrix operand order repaired and checked by reading, not compiled, not run on hardware
//
#include <hip/hip_runtime.h>
#include <stdint.h>

#define D 512
#define LSTRIDE 520            // LDS row stride in bf16 elems (pad 8 -> conflict-free b128 loads)
#define BM 64                  // batch rows per workgroup
#define BLOCK 256              // 8 wave32

typedef __attribute__((ext_vector_type(16))) __bf16        bf16x16;
typedef __attribute__((ext_vector_type(8)))  float         f32x8;
typedef __attribute__((ext_vector_type(4)))  unsigned int  u32x4;

union Frag { bf16x16 v; u32x4 q[2]; };

__device__ __forceinline__ unsigned short f2bf(float f) {
  unsigned u = __float_as_uint(f);
  u += 0x7FFFu + ((u >> 16) & 1u);            // round-to-nearest-even
  return (unsigned short)(u >> 16);
}
__device__ __forceinline__ float bf2f(unsigned short s) {
  return __uint_as_float(((unsigned)s) << 16);
}

// gfx1250 has hardware V_TANH_F32 (TRANS pipe, co-executes with WMMA).
__device__ __forceinline__ float fast_tanh(float x) {
#if __has_builtin(__builtin_amdgcn_tanhf)
  return __builtin_amdgcn_tanhf(x);
#elif __has_builtin(__builtin_amdgcn_tanh_f32)
  return __builtin_amdgcn_tanh_f32(x);
#else
  return tanhf(x);
#endif
}
// sigmoid(x) == 0.5*tanh(x/2) + 0.5 exactly
__device__ __forceinline__ float fast_sigmoid(float x) {
  return fmaf(0.5f, fast_tanh(0.5f * x), 0.5f);
}

// A fragment: 16x32 bf16 tile, rows r0..r0+15 of LDS activation (stride LSTRIDE).
// lane(l,h): row=r0+l, elems 0..7 -> K=k0+8h.., elems 8..15 -> K=k0+16+8h..
__device__ __forceinline__ bf16x16 load_a_frag(const unsigned short* base, int r0, int k0, int l, int h) {
  const unsigned short* p = base + (r0 + l) * LSTRIDE + k0 + 8 * h;
  Frag f;
  f.q[0] = *(const u32x4*)(p);
  f.q[1] = *(const u32x4*)(p + 16);
  return f.v;
}
// B fragment: 32x16 tile of W (K x N). Wt is W^T stored row-major [N][K] bf16, so
// lane(l,h): col=j0+l, elems 0..7 -> K=k0+8h.., elems 8..15 -> K=k0+16+8h..
__device__ __forceinline__ bf16x16 load_b_frag(const unsigned short* __restrict__ Wt, int j0, int k0, int l, int h) {
  const unsigned short* p = Wt + (j0 + l) * D + k0 + 8 * h;
  Frag f;
  f.q[0] = *(const u32x4*)(p);
  f.q[1] = *(const u32x4*)(p + 16);
  return f.v;
}
__device__ __forceinline__ f32x8 wmma_bf16(bf16x16 a, bf16x16 b, f32x8 c) {
  return __builtin_amdgcn_wmma_f32_16x16x32_bf16(false, a, false, b, (short)0, c, false, false);
}

// dst[64][D] = act(src[64][D] @ W + bias). 2-wide column-tile blocking:
// each A fragment feeds 2 B tiles -> 1:1 ds_load_b128 : v_wmma.
template<bool ACT_TANH, bool DUAL>
__device__ __forceinline__ void gemm_strip(
    const unsigned short* src, const unsigned short* __restrict__ Wt,
    const float* __restrict__ bias,
    unsigned short* dst, unsigned short* dst2, int w, int l, int h)
{
  const f32x8 vzero = {0.f,0.f,0.f,0.f,0.f,0.f,0.f,0.f};
  for (int cp = 0; cp < 2; ++cp) {
    const int j0 = w * 64 + cp * 32;
    f32x8 acc0[4], acc1[4];
#pragma unroll
    for (int i = 0; i < 4; ++i) { acc0[i] = vzero; acc1[i] = vzero; }
    for (int k0 = 0; k0 < D; k0 += 32) {
      bf16x16 b0 = load_b_frag(Wt, j0,      k0, l, h);
      bf16x16 b1 = load_b_frag(Wt, j0 + 16, k0, l, h);
#pragma unroll
      for (int rt = 0; rt < 4; ++rt) {
        bf16x16 a = load_a_frag(src, rt * 16, k0, l, h);
        acc0[rt] = wmma_bf16(a, b0, acc0[rt]);
        acc1[rt] = wmma_bf16(a, b1, acc1[rt]);
      }
    }
#pragma unroll
    for (int sub = 0; sub < 2; ++sub) {
      const int col = j0 + sub * 16 + l;
      const float bb = bias[col];
      f32x8* acc = sub ? acc1 : acc0;
#pragma unroll
      for (int rt = 0; rt < 4; ++rt) {
#pragma unroll
        for (int v = 0; v < 8; ++v) {
          const int row = rt * 16 + v + 8 * h;
          float z = acc[rt][v] + bb;
          if (ACT_TANH) z = fast_tanh(z);
          const unsigned short bz = f2bf(z);
          dst[row * LSTRIDE + col] = bz;
          if (DUAL) dst2[row * LSTRIDE + col] = bz;
        }
      }
    }
  }
}

// ---------------- preprocessing: pack weights as bf16 W^T, fold S1==x pairs ----------------

__global__ void dgm_prep_transpose2(const float* __restrict__ Wh1, const float* __restrict__ Whd,
                                    unsigned short* __restrict__ ws) {
  int idx = blockIdx.x * blockDim.x + threadIdx.x;       // 2 * D * D
  if (idx >= 2 * D * D) return;
  int m = idx >> 18;
  int o = idx & (D * D - 1);
  int n = o >> 9, k = o & (D - 1);
  const float* W = m ? Whd : Wh1;
  ws[(size_t)m * D * D + o] = f2bf(W[k * D + n]);
}

__global__ void dgm_prep_gates(const float* __restrict__ dgmW, unsigned short* __restrict__ ws) {
  int idx = blockIdx.x * blockDim.x + threadIdx.x;       // 24 * D * D
  if (idx >= 24 * D * D) return;
  int m = idx >> 18;
  int o = idx & (D * D - 1);
  int n = o >> 9, k = o & (D - 1);
  int ly = m >> 3, g = m & 7;
  // per layer: [WzX, WzS, WgX, WgS, WrX, WrS, WhX, WhS]; X-parts fold two matrices (S1==x)
  const int i1[8] = {0, 1, 3, 5, 6, 7, 9, 10};
  const int i2[8] = {2, -1, 4, -1, 8, -1, 11, -1};
  size_t base = (size_t)ly * 12 * D * D;
  float v = dgmW[base + (size_t)i1[g] * D * D + (size_t)k * D + n];
  if (i2[g] >= 0) v += dgmW[base + (size_t)i2[g] * D * D + (size_t)k * D + n];
  ws[(size_t)(2 + m) * D * D + o] = f2bf(v);
}

__global__ void dgm_prep_bias(const float* __restrict__ dgmB, float* __restrict__ wsB) {
  int idx = blockIdx.x * blockDim.x + threadIdx.x;       // 3 * 4 * D
  if (idx >= 3 * 4 * D) return;
  int ly = idx / (4 * D);
  int g = (idx / D) & 3;
  int n = idx & (D - 1);
  float v = 0.f;
#pragma unroll
  for (int j = 0; j < 3; ++j) v += dgmB[(ly * 12 + 3 * g + j) * D + n];
  wsB[idx] = v;
}

// ---------------- fused DGM forward: whole network per 64-row batch tile ----------------

__global__ __launch_bounds__(BLOCK, 1)
void dgm_fused(const float* __restrict__ X, const float* __restrict__ T,
               const float* __restrict__ W_in, const float* __restrict__ b_in,
               const float* __restrict__ b_h1, const float* __restrict__ b_hd,
               const unsigned short* __restrict__ wsW, const float* __restrict__ wsB,
               const float* __restrict__ W_out, const float* __restrict__ b_out,
               float* __restrict__ out)
{
  extern __shared__ unsigned short smem[];
  unsigned short* bufX = smem;                  // x (== S1, constant across layers)
  unsigned short* bufA = bufX + BM * LSTRIDE;   // S ping
  unsigned short* bufB = bufA + BM * LSTRIDE;   // S pong
  unsigned short* bufR = bufB + BM * LSTRIDE;   // S * R

  const int tid = threadIdx.x;
  const int lane = tid & 31;
  const int w = tid >> 5;
  const int l = lane & 15;
  const int h = lane >> 4;
  const int rbase = blockIdx.x * BM;
  const f32x8 vzero = {0.f,0.f,0.f,0.f,0.f,0.f,0.f,0.f};

  // Stage 0: h0 = tanh([X t] @ W_in + b_in)  (K=3: plain VALU) -> bufA
  // one row per thread (4 threads/row, 128 cols each): X/t loaded once per thread
  {
    const int row = tid & 63;
    const int c0 = (tid >> 6) * 128;
    const int gr = rbase + row;
    const float x0 = X[gr * 2 + 0], x1 = X[gr * 2 + 1], tt = T[gr];
    for (int c = c0; c < c0 + 128; ++c) {
      float v = fast_tanh(x0 * W_in[c] + x1 * W_in[D + c] + tt * W_in[2 * D + c] + b_in[c]);
      bufA[row * LSTRIDE + c] = f2bf(v);
    }
  }
  __syncthreads();

  const unsigned short* MW0 = wsW;  // matrix i at wsW + i*D*D
  // Stage 1: h1 = tanh(h0 @ W_h1 + b_h1) -> bufB
  gemm_strip<true, false>(bufA, MW0 + 0 * (size_t)D * D, b_h1, bufB, nullptr, w, l, h);
  __syncthreads();
  // Stage 2: x = h1 @ W_hd + b_hd -> bufX, and S0 = x -> bufA
  gemm_strip<false, true>(bufB, MW0 + 1 * (size_t)D * D, b_hd, bufX, bufA, w, l, h);
  __syncthreads();

  for (int ly = 0; ly < 3; ++ly) {
    unsigned short* Sp = (ly & 1) ? bufB : bufA;   // current S
    unsigned short* Sn = (ly & 1) ? bufA : bufB;   // next S
    const unsigned short* Wm = wsW + (size_t)(2 + ly * 8) * D * D;
    const unsigned short* WzX = Wm + 0 * (size_t)D * D;
    const unsigned short* WzS = Wm + 1 * (size_t)D * D;
    const unsigned short* WgX = Wm + 2 * (size_t)D * D;
    const unsigned short* WgS = Wm + 3 * (size_t)D * D;
    const unsigned short* WrX = Wm + 4 * (size_t)D * D;
    const unsigned short* WrS = Wm + 5 * (size_t)D * D;
    const unsigned short* WhX = Wm + 6 * (size_t)D * D;
    const unsigned short* WhS = Wm + 7 * (size_t)D * D;
    const float* bz = wsB + (ly * 4 + 0) * D;
    const float* bg = wsB + (ly * 4 + 1) * D;
    const float* br = wsB + (ly * 4 + 2) * D;
    const float* bh = wsB + (ly * 4 + 3) * D;

    // Phase R: R = sigmoid(x@WrX + S@WrS + br); bufR = S * R.
    // 2-wide column blocking: A fragments (x,S) shared by 2 column tiles.
    for (int cp = 0; cp < 2; ++cp) {
      const int j0 = w * 64 + cp * 32;
      f32x8 acc0[4], acc1[4];
#pragma unroll
      for (int i = 0; i < 4; ++i) { acc0[i] = vzero; acc1[i] = vzero; }
      for (int k0 = 0; k0 < D; k0 += 32) {
        bf16x16 bx0 = load_b_frag(WrX, j0,      k0, l, h);
        bf16x16 bx1 = load_b_frag(WrX, j0 + 16, k0, l, h);
        bf16x16 bs0 = load_b_frag(WrS, j0,      k0, l, h);
        bf16x16 bs1 = load_b_frag(WrS, j0 + 16, k0, l, h);
#pragma unroll
        for (int rt = 0; rt < 4; ++rt) {
          bf16x16 aX = load_a_frag(bufX, rt * 16, k0, l, h);
          bf16x16 aS = load_a_frag(Sp,   rt * 16, k0, l, h);
          acc0[rt] = wmma_bf16(aX, bx0, acc0[rt]);
          acc0[rt] = wmma_bf16(aS, bs0, acc0[rt]);
          acc1[rt] = wmma_bf16(aX, bx1, acc1[rt]);
          acc1[rt] = wmma_bf16(aS, bs1, acc1[rt]);
        }
      }
#pragma unroll
      for (int sub = 0; sub < 2; ++sub) {
        const int col = j0 + sub * 16 + l;
        const float bb = br[col];
        f32x8* acc = sub ? acc1 : acc0;
#pragma unroll
        for (int rt = 0; rt < 4; ++rt) {
#pragma unroll
          for (int v = 0; v < 8; ++v) {
            const int row = rt * 16 + v + 8 * h;
            float rv = fast_sigmoid(acc[rt][v] + bb);
            float sv = bf2f(Sp[row * LSTRIDE + col]);
            bufR[row * LSTRIDE + col] = f2bf(sv * rv);
          }
        }
      }
    }
    __syncthreads();

    // Phase ZGH: Z,G,H per 16-col tile in registers (x/S/SR fragments shared by
    // the 3 gates -> already 1:1 ds:wmma); S_new = (1-G)*H + Z*S -> Sn
    for (int ct = 0; ct < 4; ++ct) {
      const int j0 = w * 64 + ct * 16;
      f32x8 aZ[4], aG[4], aH[4];
#pragma unroll
      for (int i = 0; i < 4; ++i) { aZ[i] = vzero; aG[i] = vzero; aH[i] = vzero; }
      for (int k0 = 0; k0 < D; k0 += 32) {
        bf16x16 bzx = load_b_frag(WzX, j0, k0, l, h);
        bf16x16 bzs = load_b_frag(WzS, j0, k0, l, h);
        bf16x16 bgx = load_b_frag(WgX, j0, k0, l, h);
        bf16x16 bgs = load_b_frag(WgS, j0, k0, l, h);
        bf16x16 bhx = load_b_frag(WhX, j0, k0, l, h);
        bf16x16 bhs = load_b_frag(WhS, j0, k0, l, h);
#pragma unroll
        for (int rt = 0; rt < 4; ++rt) {
          bf16x16 ax = load_a_frag(bufX, rt * 16, k0, l, h);
          bf16x16 as = load_a_frag(Sp,   rt * 16, k0, l, h);
          bf16x16 ar = load_a_frag(bufR, rt * 16, k0, l, h);
          aZ[rt] = wmma_bf16(ax, bzx, aZ[rt]);
          aZ[rt] = wmma_bf16(as, bzs, aZ[rt]);
          aG[rt] = wmma_bf16(ax, bgx, aG[rt]);
          aG[rt] = wmma_bf16(as, bgs, aG[rt]);
          aH[rt] = wmma_bf16(ax, bhx, aH[rt]);
          aH[rt] = wmma_bf16(ar, bhs, aH[rt]);
        }
      }
      const float bzv = bz[j0 + l], bgv = bg[j0 + l], bhv = bh[j0 + l];
#pragma unroll
      for (int rt = 0; rt < 4; ++rt) {
#pragma unroll
        for (int v = 0; v < 8; ++v) {
          const int row = rt * 16 + v + 8 * h;
          const int col = j0 + l;
          float zz = fast_sigmoid(aZ[rt][v] + bzv);
          float gg = fast_sigmoid(aG[rt][v] + bgv);
          float hh = fast_tanh(aH[rt][v] + bhv);
          float so = bf2f(Sp[row * LSTRIDE + col]);
          Sn[row * LSTRIDE + col] = f2bf((1.0f - gg) * hh + zz * so);
        }
      }
    }
    __syncthreads();
  }

  // Output head: out = S @ W_out + b_out   (final S in bufB after 3 layers)
  {
    const int row = tid >> 2;
    const int q = tid & 3;
    float s = 0.f;
    const int c0 = q * 128;
    for (int c = c0; c < c0 + 128; ++c)
      s += bf2f(bufB[row * LSTRIDE + c]) * W_out[c];
    s += __shfl_xor(s, 1);
    s += __shfl_xor(s, 2);
    if (q == 0) out[rbase + row] = s + b_out[0];
  }
}

extern "C" void kernel_launch(void* const* d_in, const int* in_sizes, int n_in,
                              void* d_out, int out_size, void* d_ws, size_t ws_size,
                              hipStream_t stream) {
  (void)n_in; (void)out_size; (void)ws_size;
  const float* X     = (const float*)d_in[0];
  const float* T     = (const float*)d_in[1];
  const float* W_in  = (const float*)d_in[2];
  const float* b_in  = (const float*)d_in[3];
  const float* W_h1  = (const float*)d_in[4];
  const float* b_h1  = (const float*)d_in[5];
  const float* W_hd  = (const float*)d_in[6];
  const float* b_hd  = (const float*)d_in[7];
  const float* dgmW  = (const float*)d_in[8];
  const float* dgmB  = (const float*)d_in[9];
  const float* W_out = (const float*)d_in[10];
  const float* b_out = (const float*)d_in[11];
  float* out = (float*)d_out;

  unsigned short* wsW = (unsigned short*)d_ws;                       // 26 packed bf16 W^T matrices
  float* wsB = (float*)((char*)d_ws + (size_t)26 * D * D * sizeof(unsigned short));

  const int B = in_sizes[0] / 2;

  dgm_prep_transpose2<<<(2 * D * D + 255) / 256, 256, 0, stream>>>(W_h1, W_hd, wsW);
  dgm_prep_gates<<<(24 * D * D + 255) / 256, 256, 0, stream>>>(dgmW, wsW);
  dgm_prep_bias<<<(3 * 4 * D + 255) / 256, 256, 0, stream>>>(dgmB, wsB);

  const size_t lds_bytes = (size_t)4 * BM * LSTRIDE * sizeof(unsigned short);  // 266,240 B < 320 KB
  (void)hipFuncSetAttribute((const void*)dgm_fused,
                            hipFuncAttributeMaxDynamicSharedMemorySize, (int)lds_bytes);
  dgm_fused<<<B / BM, BLOCK, lds_bytes, stream>>>(X, T, W_in, b_in, b_h1, b_hd,
                                                  wsW, wsB, W_out, b_out, out);
}
